// PCAGProbabilisticModule_27101243638412
// MI455X (gfx1250) — matrix operand order repaired and checked by reading, not compiled, not run on hardware
//
#include <hip/hip_runtime.h>

typedef __bf16 bf16_t;
typedef __attribute__((ext_vector_type(16))) __bf16 bf16x16;
typedef __attribute__((ext_vector_type(8)))  float  floatx8;

#define DIM  768
#define NBB  32
#define NS   256
#define NM   3
#define NH   1024
#define ROWS (NBB*NS)   // 8192
#define KDIM2 (NM*DIM)  // 2304

// ---------------------------------------------------------------------------
// WMMA fragment loaders (CDNA5 wave32 layouts, CDNA5 ISA 7.12.2)
// ---------------------------------------------------------------------------
__device__ inline bf16x16 load_a_frag(const bf16_t* As, int lane) {
  int half = lane >> 4, r = lane & 15;
  bf16x16 a;
#pragma unroll
  for (int v = 0; v < 8; ++v) {
    int k = ((v < 4) ? (v * 2) : (16 + (v - 4) * 2)) + half * 8;
    a[2 * v]     = As[r * 32 + k];
    a[2 * v + 1] = As[r * 32 + k + 1];
  }
  return a;
}

__device__ inline bf16x16 load_b_frag(const bf16_t* Bt, int lane) {
  int half = lane >> 4, n = lane & 15;
  bf16x16 b;
#pragma unroll
  for (int t = 0; t < 16; ++t) b[t] = Bt[n * 32 + half * 8 + (t < 8 ? t : 8 + t)];
  return b;
}

__device__ inline floatx8 wmma_bf16(bf16x16 a, bf16x16 b, floatx8 c) {
  return __builtin_amdgcn_wmma_f32_16x16x32_bf16(false, a, false, b, (short)0, c,
                                                 false, false);
}

__device__ inline float block_sum(float v, float* red, int tid) {
  red[tid] = v; __syncthreads();
  for (int st = 128; st > 0; st >>= 1) {
    if (tid < st) red[tid] += red[tid + st];
    __syncthreads();
  }
  float r = red[0]; __syncthreads();
  return r;
}

__device__ inline float block_max(float v, float* red, int tid) {
  red[tid] = v; __syncthreads();
  for (int st = 128; st > 0; st >>= 1) {
    if (tid < st) red[tid] = fmaxf(red[tid], red[tid + st]);
    __syncthreads();
  }
  float r = red[0]; __syncthreads();
  return r;
}

__device__ inline float sigm(float x) { return 1.0f / (1.0f + expf(-x)); }

__device__ inline float dot4(float4 a, float4 b) {
  return a.x * b.x + a.y * b.y + a.z * b.z + a.w * b.w;
}

// ---------------------------------------------------------------------------
// 0a) fp32 -> bf16 (row-major copy, for feats)
// ---------------------------------------------------------------------------
__global__ void k_cvt_bf16(const float* __restrict__ src, bf16_t* __restrict__ dst,
                           int n) {
  int i = blockIdx.x * 256 + threadIdx.x;
  if (i < n) dst[i] = (bf16_t)src[i];
}

// ---------------------------------------------------------------------------
// 0b) fp32 [Dn][En] -> bf16 transposed [En][Dn], tiled through LDS.
//     grid (Dn/32, En/32, nmat), block 256 (32x8).
// ---------------------------------------------------------------------------
__global__ void k_cvtT(const float* __restrict__ src, bf16_t* __restrict__ dst,
                       int Dn, int En) {
  __shared__ float tile[32][33];
  size_t mo = (size_t)blockIdx.z * Dn * En;
  src += mo; dst += mo;
  int d0 = blockIdx.x * 32, e0 = blockIdx.y * 32;
  int lx = threadIdx.x & 31, ly = threadIdx.x >> 5;  // 32 cols x 8 rows
#pragma unroll
  for (int r = 0; r < 32; r += 8)
    tile[ly + r][lx] = src[(size_t)(d0 + ly + r) * En + e0 + lx];
  __syncthreads();
#pragma unroll
  for (int r = 0; r < 32; r += 8)
    dst[(size_t)(e0 + ly + r) * Dn + d0 + lx] = (bf16_t)tile[lx][ly + r];
}

// ---------------------------------------------------------------------------
// 1) rank-1 precompute: u = W @ w (per modality), plus scalar constants.
// ---------------------------------------------------------------------------
__global__ void k_precompute(const float* Wq, const float* Wk,
                             const float* bq, const float* bk,
                             const float* mu_w, const float* mu_b,
                             const float* lv_w, const float* lv_b,
                             const float* pgq_w, const float* pgq_b,
                             const float* pgk_w, const float* pgk_b,
                             float* u_mu, float* u_lv, float* u_pq, float* u_pk,
                             float* consts) {
  int i = blockIdx.x, kind = blockIdx.y, tid = threadIdx.x;
  const float* W; const float* wv; float* outp;
  int j = 0;
  if (kind == 0)      { W = Wq + (size_t)i * DIM * DIM; wv = mu_w  + i * DIM; outp = u_mu + i * DIM; }
  else if (kind == 1) { W = Wq + (size_t)i * DIM * DIM; wv = lv_w  + i * DIM; outp = u_lv + i * DIM; }
  else if (kind == 2) { W = Wq + (size_t)i * DIM * DIM; wv = pgq_w + i * DIM; outp = u_pq + i * DIM; }
  else { j = kind - 3; W = Wk + (size_t)j * DIM * DIM; wv = pgk_w + i * DIM;
         outp = u_pk + (size_t)(i * NM + j) * DIM; }
  for (int d = tid; d < DIM; d += 256) {
    float acc = 0.f;
    const float4* row = (const float4*)(W + (size_t)d * DIM);
    const float4* wv4 = (const float4*)wv;
    for (int e = 0; e < DIM / 4; ++e) acc += dot4(row[e], wv4[e]);
    outp[d] = acc;
  }
  if (tid == 0) {
    const float* bb = (kind < 3) ? (bq + i * DIM) : (bk + j * DIM);
    float c = 0.f;
    for (int e = 0; e < DIM; ++e) c += bb[e] * wv[e];
    if (kind == 0)      consts[i]              = c + mu_b[i];
    else if (kind == 1) consts[3 + i]          = c + lv_b[i];
    else if (kind == 2) consts[6 + i]          = c + pgq_b[i];
    else                consts[9 + i * NM + j] = c + pgk_b[i];
  }
}

// ---------------------------------------------------------------------------
// 2) K/V projections (the only full GEMMs): bf16 WMMA, fp32 accumulate.
//    Double-buffered LDS, straight-line register staging, pre-transposed W.
//    grid (64, 12, 6), block 256 = 8 waves (4x2), each wave owns 32x32.
// ---------------------------------------------------------------------------
__global__ void k_gemm_kv(const bf16_t* __restrict__ featsbf,
                          const bf16_t* __restrict__ WkT,
                          const bf16_t* __restrict__ WvT,
                          const float* __restrict__ bk,
                          const float* __restrict__ bv,
                          float* __restrict__ Kf, float* __restrict__ Vf) {
  int mt = blockIdx.x, nt = blockIdx.y, z = blockIdx.z;
  int which = z & 1, m = z >> 1;
  const bf16_t* A  = featsbf + (size_t)m * ROWS * DIM;
  const bf16_t* WT = (which ? WvT : WkT) + (size_t)m * DIM * DIM;  // [E][D]
  const float* bias = (which ? bv : bk) + m * DIM;
  float* Out = (which ? Vf : Kf) + (size_t)m * ROWS * DIM;

  __shared__ __align__(16) bf16_t As[2][128 * 32];
  __shared__ __align__(16) bf16_t Bt[2][64 * 32];

  int tid = threadIdx.x, lane = tid & 31, wave = tid >> 5;
  int wm = wave & 3, wn = wave >> 2;
  int row0 = mt * 128, col0 = nt * 64;
  int ar = tid >> 2, akk = (tid & 3) * 8;  // 64 rows x 4 chunks of 8 bf16

  floatx8 acc[2][2];
#pragma unroll
  for (int a = 0; a < 2; ++a)
#pragma unroll
    for (int b = 0; b < 2; ++b)
      acc[a][b] = (floatx8){0.f, 0.f, 0.f, 0.f, 0.f, 0.f, 0.f, 0.f};

  float4 ra0, ra1, rb;
  const int NIT = DIM / 32;

  // prologue: tile 0
  ra0 = *(const float4*)(A + (size_t)(row0 + ar) * DIM + akk);
  ra1 = *(const float4*)(A + (size_t)(row0 + 64 + ar) * DIM + akk);
  rb  = *(const float4*)(WT + (size_t)(col0 + ar) * DIM + akk);
  *(float4*)&As[0][ar * 32 + akk] = ra0;
  *(float4*)&As[0][(64 + ar) * 32 + akk] = ra1;
  *(float4*)&Bt[0][ar * 32 + akk] = rb;
  __syncthreads();

  for (int it = 0; it < NIT; ++it) {
    int cur = it & 1;
    if (it + 1 < NIT) {  // issue next tile's global loads before the math
      int k0 = (it + 1) * 32;
      ra0 = *(const float4*)(A + (size_t)(row0 + ar) * DIM + k0 + akk);
      ra1 = *(const float4*)(A + (size_t)(row0 + 64 + ar) * DIM + k0 + akk);
      rb  = *(const float4*)(WT + (size_t)(col0 + ar) * DIM + k0 + akk);
    }
    bf16x16 af[2], bfr[2];
    af[0]  = load_a_frag(&As[cur][(wm * 32 + 0) * 32], lane);
    af[1]  = load_a_frag(&As[cur][(wm * 32 + 16) * 32], lane);
    bfr[0] = load_b_frag(&Bt[cur][(wn * 32 + 0) * 32], lane);
    bfr[1] = load_b_frag(&Bt[cur][(wn * 32 + 16) * 32], lane);
#pragma unroll
    for (int fm = 0; fm < 2; ++fm)
#pragma unroll
      for (int fn = 0; fn < 2; ++fn)
        acc[fm][fn] = wmma_bf16(af[fm], bfr[fn], acc[fm][fn]);
    if (it + 1 < NIT) {
      int nb = cur ^ 1;
      *(float4*)&As[nb][ar * 32 + akk] = ra0;
      *(float4*)&As[nb][(64 + ar) * 32 + akk] = ra1;
      *(float4*)&Bt[nb][ar * 32 + akk] = rb;
    }
    __syncthreads();
  }

  int half = lane >> 4, nl = lane & 15;
#pragma unroll
  for (int fm = 0; fm < 2; ++fm)
#pragma unroll
    for (int fn = 0; fn < 2; ++fn)
#pragma unroll
      for (int r = 0; r < 8; ++r) {
        int row = row0 + wm * 32 + fm * 16 + r + 8 * half;
        int col = col0 + wn * 32 + fn * 16 + nl;
        Out[(size_t)row * DIM + col] = acc[fm][fn][r] + bias[col];
      }
}

// ---------------------------------------------------------------------------
// 3) Q projection, s=0 rows only (contiguous via transposed bf16 Wq)
// ---------------------------------------------------------------------------
__global__ void k_qrow0(const float* __restrict__ feats,
                        const bf16_t* __restrict__ WqT,
                        const float* __restrict__ bq, float* __restrict__ Q0) {
  int idx = blockIdx.x * 256 + threadIdx.x;  // 73728 total
  int e = idx % DIM, ib = idx / DIM;
  int b = ib % NBB, i = ib / NBB;
  const float* fr = feats + ((size_t)(i * NBB + b) * NS) * DIM;  // s = 0
  const bf16_t* wc = WqT + (size_t)i * DIM * DIM + (size_t)e * DIM;
  float acc = bq[i * DIM + e];
  for (int d = 0; d < DIM; ++d) acc += fr[d] * (float)wc[d];
  Q0[idx] = acc;
}

// ---------------------------------------------------------------------------
// 4) gates + KL via rank-1 dots. grid (3, 32), block 256 (thread = pos s)
// ---------------------------------------------------------------------------
__global__ void k_gates_kl(const float* __restrict__ feats,
                           const float* __restrict__ u_mu, const float* __restrict__ u_lv,
                           const float* __restrict__ u_pq, const float* __restrict__ u_pk,
                           const float* __restrict__ consts, const float* __restrict__ dyn,
                           const float* __restrict__ eps,
                           float* __restrict__ pkf, float* __restrict__ pq0,
                           float* __restrict__ klpart) {
  int i = blockIdx.x, b = blockIdx.y, s = threadIdx.x;
  __shared__ float red[256];
  const float4* fr4 = (const float4*)(feats + ((size_t)(i * NBB + b) * NS + s) * DIM);
  const float4* um4 = (const float4*)(u_mu + i * DIM);
  const float4* ul4 = (const float4*)(u_lv + i * DIM);
  float mu = consts[i], lv = consts[3 + i];
  for (int d = 0; d < DIM / 4; ++d) {
    float4 f = fr4[d];
    mu += dot4(f, um4[d]);
    lv += dot4(f, ul4[d]);
  }
  float stdv = expf(0.5f * lv);
  float klt = 1.0f + lv - mu * mu - expf(lv);
  float tot = block_sum(klt, red, s);
  if (s == 0) klpart[i * NBB + b] = -0.5f * tot * (float)(NM - 1);

  float pq_base = 0.f;
  if (s == 0) {
    const float4* up4 = (const float4*)(u_pq + i * DIM);
    float d0 = consts[6 + i];
    for (int d = 0; d < DIM / 4; ++d) d0 += dot4(fr4[d], up4[d]);
    pq_base = sigm(d0);
  }
  float dynv = dyn[0];
  for (int cnt = 0; cnt < NM - 1; ++cnt) {
    int j = cnt + (cnt >= i ? 1 : 0);
    float e = eps[((size_t)(i * (NM - 1) + cnt) * NBB + b) * NS + s];
    float g = sigm(mu + dynv * stdv * e);
    const float4* fj4 = (const float4*)(feats + ((size_t)(j * NBB + b) * NS + s) * DIM);
    const float4* uk4 = (const float4*)(u_pk + (size_t)(i * NM + j) * DIM);
    float kd = consts[9 + i * NM + j];
    for (int d = 0; d < DIM / 4; ++d) kd += dot4(fj4[d], uk4[d]);
    pkf[((size_t)(i * (NM - 1) + cnt) * NBB + b) * NS + s] = sigm(kd) * g;
    if (s == 0) pq0[(i * (NM - 1) + cnt) * NBB + b] = pq_base * g;
  }
}

__global__ void k_kl_reduce(const float* __restrict__ klpart, float* __restrict__ out) {
  if (blockIdx.x == 0 && threadIdx.x == 0) {
    float sum = 0.f;
    for (int t = 0; t < NM * NBB; ++t) sum += klpart[t];  // fixed order: deterministic
    out[NBB * NH] = sum;
  }
}

// ---------------------------------------------------------------------------
// 5) attention, query row 0 only. grid (3, 32), block 256 (thread = key k)
// ---------------------------------------------------------------------------
__global__ void k_attn_row0(const float* __restrict__ Kf, const float* __restrict__ Vf,
                            const float* __restrict__ Q0, const float* __restrict__ pkf,
                            const float* __restrict__ pq0, float* __restrict__ agg0) {
  int i = blockIdx.x, b = blockIdx.y, tid = threadIdx.x;
  __shared__ __align__(16) float q[DIM];
  __shared__ float wgt[NS];
  __shared__ float red[256];
  for (int t = tid; t < DIM; t += 256) q[t] = Q0[(size_t)(i * NBB + b) * DIM + t];
  const float scale = rsqrtf((float)DIM);
  float a0 = 0.f, a1 = 0.f, a2 = 0.f;
  for (int cnt = 0; cnt < NM - 1; ++cnt) {
    int j = cnt + (cnt >= i ? 1 : 0);
    __syncthreads();
    const float4* K4 = (const float4*)(Kf + ((size_t)j * ROWS + b * NS + tid) * DIM);
    const float4* q4 = (const float4*)q;
    float d = 0.f;
    for (int e = 0; e < DIM / 4; ++e) d += dot4(q4[e], K4[e]);
    float sc = d * scale * pq0[(i * (NM - 1) + cnt) * NBB + b] *
               pkf[((size_t)(i * (NM - 1) + cnt) * NBB + b) * NS + tid];
    float mx = block_max(sc, red, tid);
    float ex = expf(sc - mx);
    float sm = block_sum(ex, red, tid);
    wgt[tid] = ex / sm;
    __syncthreads();
    const float* Vb = Vf + ((size_t)j * ROWS + b * NS) * DIM;
    float t0 = 0.f, t1 = 0.f, t2 = 0.f;
    for (int k = 0; k < NS; ++k) {
      float wk = wgt[k];
      const float* vr = Vb + (size_t)k * DIM;
      t0 += wk * vr[tid];
      t1 += wk * vr[tid + 256];
      t2 += wk * vr[tid + 512];
    }
    a0 += t0; a1 += t1; a2 += t2;
  }
  float* op = agg0 + (size_t)(i * NBB + b) * DIM;
  op[tid] = a0; op[tid + 256] = a1; op[tid + 512] = a2;
}

// ---------------------------------------------------------------------------
// 6) CAG GEMMs on the 32 live rows (WMMA). grid (3, 12), 8 waves = 2x4.
// ---------------------------------------------------------------------------
__global__ void k_cag(const float* __restrict__ agg0, const float* __restrict__ Q0,
                      const bf16_t* __restrict__ WET, const bf16_t* __restrict__ WhT,
                      const bf16_t* __restrict__ WqcT,
                      const float* __restrict__ WEb, const float* __restrict__ Whb,
                      const float* __restrict__ Wqcb,
                      float* __restrict__ Ebuf, float* __restrict__ Gbuf) {
  int i = blockIdx.x, nt = blockIdx.y;
  int tid = threadIdx.x, lane = tid & 31, wave = tid >> 5;
  int wm = wave & 1, wn = wave >> 1;
  int col0 = nt * 64;

  __shared__ __align__(16) bf16_t aggA[32 * 32];
  __shared__ __align__(16) bf16_t qA[32 * 32];
  __shared__ __align__(16) bf16_t BtE[64 * 32];
  __shared__ __align__(16) bf16_t BtH[64 * 32];
  __shared__ __align__(16) bf16_t BtQ[64 * 32];

  const bf16_t* We = WET  + (size_t)i * DIM * DIM;  // [E][D]
  const bf16_t* Wh = WhT  + (size_t)i * DIM * DIM;
  const bf16_t* Wc = WqcT + (size_t)i * DIM * DIM;

  floatx8 accE = (floatx8){0.f, 0.f, 0.f, 0.f, 0.f, 0.f, 0.f, 0.f};
  floatx8 accG = accE;

  int sr = tid >> 3, skk = (tid & 7) * 4;   // A: 32 rows x 8 chunks of 4
  int br = tid >> 2, bkk = (tid & 3) * 8;   // B: 64 rows x 4 chunks of 8

  for (int k0 = 0; k0 < DIM; k0 += 32) {
    float4 va = *(const float4*)(agg0 + (size_t)(i * NBB + sr) * DIM + k0 + skk);
    float4 vq = *(const float4*)(Q0 + (size_t)(i * NBB + sr) * DIM + k0 + skk);
    bf16_t* pa = &aggA[sr * 32 + skk];
    pa[0] = (bf16_t)va.x; pa[1] = (bf16_t)va.y; pa[2] = (bf16_t)va.z; pa[3] = (bf16_t)va.w;
    bf16_t* pq = &qA[sr * 32 + skk];
    pq[0] = (bf16_t)vq.x; pq[1] = (bf16_t)vq.y; pq[2] = (bf16_t)vq.z; pq[3] = (bf16_t)vq.w;
    *(float4*)&BtE[br * 32 + bkk] = *(const float4*)(We + (size_t)(col0 + br) * DIM + k0 + bkk);
    *(float4*)&BtH[br * 32 + bkk] = *(const float4*)(Wh + (size_t)(col0 + br) * DIM + k0 + bkk);
    *(float4*)&BtQ[br * 32 + bkk] = *(const float4*)(Wc + (size_t)(col0 + br) * DIM + k0 + bkk);
    __syncthreads();
    bf16x16 aA = load_a_frag(aggA + (wm * 16) * 32, lane);
    bf16x16 aQ = load_a_frag(qA + (wm * 16) * 32, lane);
    bf16x16 bE = load_b_frag(BtE + (wn * 16) * 32, lane);
    bf16x16 bH = load_b_frag(BtH + (wn * 16) * 32, lane);
    bf16x16 bQ = load_b_frag(BtQ + (wn * 16) * 32, lane);
    accE = wmma_bf16(aA, bE, accE);
    accG = wmma_bf16(aA, bH, accG);
    accG = wmma_bf16(aQ, bQ, accG);
    __syncthreads();
  }

  int half = lane >> 4, nl = lane & 15;
#pragma unroll
  for (int r = 0; r < 8; ++r) {
    int row = wm * 16 + r + 8 * half;  // b index (0..31)
    int col = col0 + wn * 16 + nl;
    float e = accE[r] + WEb[i * DIM + col];
    float g = accG[r] + Whb[i * DIM + col] + Wqcb[i * DIM + col];
    Ebuf[(size_t)(i * NBB + row) * DIM + col] = fmaxf(e, 0.f);
    Gbuf[(size_t)(i * NBB + row) * DIM + col] = fmaxf(g, 0.f);
  }
}

// ---------------------------------------------------------------------------
// 7) LayerNorm(E) * LayerNorm(G) -> fused (bf16). grid (3, 32), block 256.
// ---------------------------------------------------------------------------
__global__ void k_lnmul(const float* __restrict__ Ebuf, const float* __restrict__ Gbuf,
                        const float* __restrict__ lnEg, const float* __restrict__ lnEb,
                        const float* __restrict__ lnGg, const float* __restrict__ lnGb,
                        bf16_t* __restrict__ fusedbf) {
  int i = blockIdx.x, b = blockIdx.y, tid = threadIdx.x;
  __shared__ float red[256];
  const float* Er = Ebuf + (size_t)(i * NBB + b) * DIM;
  const float* Gr = Gbuf + (size_t)(i * NBB + b) * DIM;
  float e0 = Er[tid], e1 = Er[tid + 256], e2 = Er[tid + 512];
  float g0 = Gr[tid], g1 = Gr[tid + 256], g2 = Gr[tid + 512];
  float sE = block_sum(e0 + e1 + e2, red, tid) * (1.0f / DIM);
  float qE = block_sum(e0 * e0 + e1 * e1 + e2 * e2, red, tid) * (1.0f / DIM);
  float sG = block_sum(g0 + g1 + g2, red, tid) * (1.0f / DIM);
  float qG = block_sum(g0 * g0 + g1 * g1 + g2 * g2, red, tid) * (1.0f / DIM);
  float rE = rsqrtf(qE - sE * sE + 1e-5f);
  float rG = rsqrtf(qG - sG * sG + 1e-5f);
  bf16_t* fo = fusedbf + (size_t)b * KDIM2 + i * DIM;
  const float* eg = lnEg + i * DIM; const float* eb = lnEb + i * DIM;
  const float* gg = lnGg + i * DIM; const float* gb = lnGb + i * DIM;
#pragma unroll
  for (int c = 0; c < 3; ++c) {
    int e = tid + c * 256;
    float ev = (c == 0 ? e0 : c == 1 ? e1 : e2);
    float gv = (c == 0 ? g0 : c == 1 ? g1 : g2);
    float le = (ev - sE) * rE * eg[e] + eb[e];
    float lg = (gv - sG) * rG * gg[e] + gb[e];
    fo[e] = (bf16_t)(le * lg);
  }
}

// ---------------------------------------------------------------------------
// 8) output GEMM: out[32,1024] = fused[32,2304] @ out_w + out_b. grid (16)
// ---------------------------------------------------------------------------
__global__ void k_out_gemm(const bf16_t* __restrict__ fusedbf,
                           const bf16_t* __restrict__ outwT,  // [NH][KDIM2]
                           const float* __restrict__ outb, float* __restrict__ out) {
  int nt = blockIdx.x;
  int tid = threadIdx.x, lane = tid & 31, wave = tid >> 5;
  int wm = wave & 1, wn = wave >> 1;
  int col0 = nt * 64;

  __shared__ __align__(16) bf16_t As[32 * 32];
  __shared__ __align__(16) bf16_t Bt[64 * 32];

  int sr = tid >> 3, skk = (tid & 7) * 4;
  int br = tid >> 2, bkk = (tid & 3) * 8;

  floatx8 acc = (floatx8){0.f, 0.f, 0.f, 0.f, 0.f, 0.f, 0.f, 0.f};
  for (int k0 = 0; k0 < KDIM2; k0 += 32) {
    *(uint2*)&As[sr * 32 + skk] =
        *(const uint2*)(fusedbf + (size_t)sr * KDIM2 + k0 + skk);
    *(float4*)&Bt[br * 32 + bkk] =
        *(const float4*)(outwT + (size_t)(col0 + br) * KDIM2 + k0 + bkk);
    __syncthreads();
    bf16x16 a = load_a_frag(As + (wm * 16) * 32, lane);
    bf16x16 b = load_b_frag(Bt + (wn * 16) * 32, lane);
    acc = wmma_bf16(a, b, acc);
    __syncthreads();
  }
  int half = lane >> 4, nl = lane & 15;
#pragma unroll
  for (int r = 0; r < 8; ++r) {
    int row = wm * 16 + r + 8 * half;  // b (0..31)
    int col = col0 + wn * 16 + nl;
    out[(size_t)row * NH + col] = acc[r] + outb[col];
  }
}

// ---------------------------------------------------------------------------
extern "C" void kernel_launch(void* const* d_in, const int* in_sizes, int n_in,
                              void* d_out, int out_size, void* d_ws, size_t ws_size,
                              hipStream_t stream) {
  (void)in_sizes; (void)n_in; (void)out_size; (void)ws_size;
  const float* feats = (const float*)d_in[0];
  const float* Wq    = (const float*)d_in[1];
  const float* bq    = (const float*)d_in[2];
  const float* Wk    = (const float*)d_in[3];
  const float* bk    = (const float*)d_in[4];
  const float* Wv    = (const float*)d_in[5];
  const float* bv    = (const float*)d_in[6];
  const float* pgq_w = (const float*)d_in[7];
  const float* pgq_b = (const float*)d_in[8];
  const float* pgk_w = (const float*)d_in[9];
  const float* pgk_b = (const float*)d_in[10];
  const float* mu_w  = (const float*)d_in[11];
  const float* mu_b  = (const float*)d_in[12];
  const float* lv_w  = (const float*)d_in[13];
  const float* lv_b  = (const float*)d_in[14];
  const float* dyn   = (const float*)d_in[15];
  const float* WEw   = (const float*)d_in[16];
  const float* WEb   = (const float*)d_in[17];
  const float* Whw   = (const float*)d_in[18];
  const float* Whb   = (const float*)d_in[19];
  const float* Wqcw  = (const float*)d_in[20];
  const float* Wqcb  = (const float*)d_in[21];
  const float* lnEg  = (const float*)d_in[22];
  const float* lnEb  = (const float*)d_in[23];
  const float* lnGg  = (const float*)d_in[24];
  const float* lnGb  = (const float*)d_in[25];
  const float* outw  = (const float*)d_in[26];
  const float* outb  = (const float*)d_in[27];
  const float* eps   = (const float*)d_in[28];
  float* out = (float*)d_out;

  char* base = (char*)d_ws; size_t off = 0;
  auto alloc = [&](size_t bytes) -> char* {
    char* p = base + off;
    off = (off + bytes + 255) & ~(size_t)255;
    return p;
  };
  bf16_t* featsbf = (bf16_t*)alloc((size_t)NM * ROWS * DIM * 2);
  bf16_t* WqT     = (bf16_t*)alloc((size_t)NM * DIM * DIM * 2);
  bf16_t* WkT     = (bf16_t*)alloc((size_t)NM * DIM * DIM * 2);
  bf16_t* WvT     = (bf16_t*)alloc((size_t)NM * DIM * DIM * 2);
  bf16_t* WET     = (bf16_t*)alloc((size_t)NM * DIM * DIM * 2);
  bf16_t* WhT     = (bf16_t*)alloc((size_t)NM * DIM * DIM * 2);
  bf16_t* WqcT    = (bf16_t*)alloc((size_t)NM * DIM * DIM * 2);
  bf16_t* outwT   = (bf16_t*)alloc((size_t)KDIM2 * NH * 2);
  float*  Kf      = (float*)alloc((size_t)NM * ROWS * DIM * 4);
  float*  Vf      = (float*)alloc((size_t)NM * ROWS * DIM * 4);
  float*  Q0f     = (float*)alloc((size_t)NM * NBB * DIM * 4);
  float*  u_mu    = (float*)alloc((size_t)NM * DIM * 4);
  float*  u_lv    = (float*)alloc((size_t)NM * DIM * 4);
  float*  u_pq    = (float*)alloc((size_t)NM * DIM * 4);
  float*  u_pk    = (float*)alloc((size_t)NM * NM * DIM * 4);
  float*  consts  = (float*)alloc(64 * 4);
  float*  pkf     = (float*)alloc((size_t)NM * (NM - 1) * NBB * NS * 4);
  float*  pq0     = (float*)alloc((size_t)NM * (NM - 1) * NBB * 4);
  float*  klpart  = (float*)alloc((size_t)NM * NBB * 4);
  float*  agg0    = (float*)alloc((size_t)NM * NBB * DIM * 4);
  float*  Ebuf    = (float*)alloc((size_t)NM * NBB * DIM * 4);
  float*  Gbuf    = (float*)alloc((size_t)NM * NBB * DIM * 4);
  bf16_t* fusedbf = (bf16_t*)alloc((size_t)NBB * KDIM2 * 2);

  // 1) rank-1 precompute
  k_precompute<<<dim3(NM, 6), 256, 0, stream>>>(Wq, Wk, bq, bk, mu_w, mu_b, lv_w,
                                                lv_b, pgq_w, pgq_b, pgk_w, pgk_b,
                                                u_mu, u_lv, u_pq, u_pk, consts);
  // 0) conversions: feats row-major, weights transposed (B side contiguous)
  {
    size_t n = (size_t)NM * ROWS * DIM;
    k_cvt_bf16<<<(int)((n + 255) / 256), 256, 0, stream>>>(feats, featsbf, (int)n);
  }
  k_cvtT<<<dim3(DIM / 32, DIM / 32, NM), 256, 0, stream>>>(Wq, WqT, DIM, DIM);
  k_cvtT<<<dim3(DIM / 32, DIM / 32, NM), 256, 0, stream>>>(Wk, WkT, DIM, DIM);
  k_cvtT<<<dim3(DIM / 32, DIM / 32, NM), 256, 0, stream>>>(Wv, WvT, DIM, DIM);
  k_cvtT<<<dim3(DIM / 32, DIM / 32, NM), 256, 0, stream>>>(WEw, WET, DIM, DIM);
  k_cvtT<<<dim3(DIM / 32, DIM / 32, NM), 256, 0, stream>>>(Whw, WhT, DIM, DIM);
  k_cvtT<<<dim3(DIM / 32, DIM / 32, NM), 256, 0, stream>>>(Wqcw, WqcT, DIM, DIM);
  k_cvtT<<<dim3(KDIM2 / 32, NH / 32, 1), 256, 0, stream>>>(outw, outwT, KDIM2, NH);

  // 2) full K/V projections (WMMA, double-buffered)
  k_gemm_kv<<<dim3(ROWS / 128, DIM / 64, 2 * NM), 256, 0, stream>>>(
      featsbf, WkT, WvT, bk, bv, Kf, Vf);
  // 3) Q row-0
  k_qrow0<<<(NM * NBB * DIM) / 256, 256, 0, stream>>>(feats, WqT, bq, Q0f);
  // 4) gates + KL partials
  k_gates_kl<<<dim3(NM, NBB), 256, 0, stream>>>(feats, u_mu, u_lv, u_pq, u_pk,
                                                consts, dyn, eps, pkf, pq0, klpart);
  k_kl_reduce<<<1, 32, 0, stream>>>(klpart, out);
  // 5) single-query attention
  k_attn_row0<<<dim3(NM, NBB), 256, 0, stream>>>(Kf, Vf, Q0f, pkf, pq0, agg0);
  // 6) CAG GEMMs (WMMA)
  k_cag<<<dim3(NM, DIM / 64), 256, 0, stream>>>(agg0, Q0f, WET, WhT, WqcT, WEb,
                                                Whb, Wqcb, Ebuf, Gbuf);
  // 7) LN*LN -> fused bf16
  k_lnmul<<<dim3(NM, NBB), 256, 0, stream>>>(Ebuf, Gbuf, lnEg, lnEb, lnGg, lnGb,
                                             fusedbf);
  // 8) output GEMM (WMMA)
  k_out_gemm<<<NH / 64, 256, 0, stream>>>(fusedbf, outwT, outb, out);
}
